// BertSelfAttention_22488448762278
// MI455X (gfx1250) — matrix operand order repaired and checked by reading, not compiled, not run on hardware
//
#include <hip/hip_runtime.h>
#include <hip/hip_bf16.h>

typedef __bf16 bf16_t;
typedef __attribute__((ext_vector_type(16))) __bf16 v16bf;
typedef __attribute__((ext_vector_type(8)))  __bf16 v8bf;
typedef __attribute__((ext_vector_type(8)))  float  v8f;

#define B_   2
#define S_   2048
#define H_   1024
#define NH_  16
#define HD_  64
#define MTOK (B_ * S_)   // 4096 tokens

// GEMM macro-tile config
#define BM 128
#define BN 64
#define BK 64

// ---------------------------------------------------------------------------
// CDNA5 primitives
// ---------------------------------------------------------------------------
__device__ __forceinline__ v8f wmma_bf16(v16bf a, v16bf b, v8f c) {
  return __builtin_amdgcn_wmma_f32_16x16x32_bf16(false, a, false, b,
                                                 (short)0, c, false, false);
}

// Async copy of 16 bytes (8 bf16) from global to LDS; tracked by ASYNCcnt.
__device__ __forceinline__ void async_copy_16B(bf16_t* lds_dst, const bf16_t* gsrc) {
  uint32_t lds_off = (uint32_t)(uintptr_t)lds_dst;      // low 32 bits = LDS offset
  unsigned long long ga = (unsigned long long)gsrc;
  asm volatile("global_load_async_to_lds_b128 %0, %1, off"
               :: "v"(lds_off), "v"(ga)
               : "memory");
}

// Wait until at most N async ops remain outstanding (loads complete in order,
// so N = #asyncs of the *next* chunk guarantees the current chunk is in LDS).
template <int N>
__device__ __forceinline__ void wait_async() {
  asm volatile("s_wait_asynccnt %0" :: "i"(N) : "memory");
}

// A fragment: 16x32 (MxK) bf16, row-major source, leading dim `ld`.
__device__ __forceinline__ v16bf load_A_frag(const bf16_t* A, int ld, int row0,
                                             int k0, int lane) {
  const int r = lane & 15, half = lane >> 4;
  const bf16_t* p = A + (size_t)(row0 + r) * ld + k0 + 8 * half;
  v8bf lo = *(const v8bf*)p;
  v8bf hi = *(const v8bf*)(p + 16);
  v16bf f;
#pragma unroll
  for (int i = 0; i < 8; ++i) { f[i] = lo[i]; f[8 + i] = hi[i]; }
  return f;
}

// B fragment: 32x16 (KxN) bf16 where B[k,n] = Wrows[n0+n][k0+k].
__device__ __forceinline__ v16bf load_B_rows(const bf16_t* W, int ld, int n0,
                                             int k0, int lane) {
  const int c = lane & 15, half = lane >> 4;
  const bf16_t* p = W + (size_t)(n0 + c) * ld + k0 + 8 * half;
  v8bf lo = *(const v8bf*)p;
  v8bf hi = *(const v8bf*)(p + 16);
  v16bf f;
#pragma unroll
  for (int i = 0; i < 8; ++i) { f[i] = lo[i]; f[8 + i] = hi[i]; }
  return f;
}

// ---------------------------------------------------------------------------
// fp32 -> bf16 conversion
// ---------------------------------------------------------------------------
__global__ void cvt_kernel(const float* __restrict__ src, bf16_t* __restrict__ dst, int n) {
  int i = blockIdx.x * blockDim.x + threadIdx.x;
  if (i < n) dst[i] = (bf16_t)src[i];
}

// ---------------------------------------------------------------------------
// LDS-staged, double-buffered GEMM core: block = 256 threads (8 waves),
// computes a BM x BN (128x64) macro-tile of  acc = X(MxK) * Wrows(NxK)^T.
// Wave w computes a 32x32 subtile: rows m0+(w&3)*32, cols n0+(w>>2)*32.
// ---------------------------------------------------------------------------
struct GemmAcc { v8f a[2][2]; };

// 6 async ops per thread per chunk: 4 for A (128x64), 2 for B (64x64)
__device__ __forceinline__ void gemm_stage(const bf16_t* __restrict__ X,
                                           const bf16_t* __restrict__ W,
                                           int m0, int n0, int kk,
                                           bf16_t* As, bf16_t* Bs, int tid) {
#pragma unroll
  for (int i = 0; i < (BM * BK) / (256 * 8); ++i) {      // 4 iters
    int f = (i * 256 + tid) * 8;
    int row = f >> 6, col = f & (BK - 1);
    async_copy_16B(&As[f], X + (size_t)(m0 + row) * H_ + kk + col);
  }
#pragma unroll
  for (int i = 0; i < (BN * BK) / (256 * 8); ++i) {      // 2 iters
    int f = (i * 256 + tid) * 8;
    int row = f >> 6, col = f & (BK - 1);
    async_copy_16B(&Bs[f], W + (size_t)(n0 + row) * H_ + kk + col);
  }
}

__device__ __forceinline__ GemmAcc gemm_core(const bf16_t* __restrict__ X,
                                             const bf16_t* __restrict__ W,
                                             int m0, int n0,
                                             bf16_t (*As)[BM * BK],
                                             bf16_t (*Bs)[BN * BK]) {
  const int tid  = threadIdx.x;
  const int lane = tid & 31;
  const int wave = tid >> 5;
  const int mi = (wave & 3) * 32;
  const int ni = (wave >> 2) * 32;

  GemmAcc acc;
#pragma unroll
  for (int tm = 0; tm < 2; ++tm)
#pragma unroll
    for (int tn = 0; tn < 2; ++tn) acc.a[tm][tn] = (v8f){};

  // prologue: stage chunk 0 into buffer 0
  gemm_stage(X, W, m0, n0, 0, As[0], Bs[0], tid);

  for (int kk = 0; kk < H_; kk += BK) {
    const int cur = (kk >> 6) & 1;
    if (kk + BK < H_) {
      gemm_stage(X, W, m0, n0, kk + BK, As[cur ^ 1], Bs[cur ^ 1], tid);
      // light prefetch of chunk k+2 toward L2
      if (kk + 2 * BK < H_)
        __builtin_prefetch(X + (size_t)(m0 + (tid & (BM - 1))) * H_ + kk + 2 * BK, 0, 0);
      wait_async<6>();          // chunk `cur` fully in LDS (in-order completion)
    } else {
      wait_async<0>();
    }
    __syncthreads();            // all waves' staging of chunk `cur` visible

    const bf16_t* Ac = As[cur];
    const bf16_t* Bc = Bs[cur];
#pragma unroll
    for (int ks = 0; ks < BK; ks += 32) {
      v16bf a0 = load_A_frag(Ac, BK, mi,      ks, lane);
      v16bf a1 = load_A_frag(Ac, BK, mi + 16, ks, lane);
      v16bf b0 = load_B_rows(Bc, BK, ni,      ks, lane);
      v16bf b1 = load_B_rows(Bc, BK, ni + 16, ks, lane);
      acc.a[0][0] = wmma_bf16(a0, b0, acc.a[0][0]);
      acc.a[0][1] = wmma_bf16(a0, b1, acc.a[0][1]);
      acc.a[1][0] = wmma_bf16(a1, b0, acc.a[1][0]);
      acc.a[1][1] = wmma_bf16(a1, b1, acc.a[1][1]);
    }
    __syncthreads();            // protect buffer reuse on the next iteration
  }
  return acc;
}

// ---------------------------------------------------------------------------
// Fused QKV projection. grid: (MTOK/BM, H/BN, 3), block 256.
// Q,K stored [B, nh, S, hd]; V stored transposed [B, nh, hd, S].
// ---------------------------------------------------------------------------
__global__ void qkv_gemm_kernel(const bf16_t* __restrict__ X,
                                const bf16_t* __restrict__ Wq,
                                const bf16_t* __restrict__ Wk,
                                const bf16_t* __restrict__ Wv,
                                const float* __restrict__ bq,
                                const float* __restrict__ bk,
                                const float* __restrict__ bv,
                                bf16_t* __restrict__ Q,
                                bf16_t* __restrict__ K,
                                bf16_t* __restrict__ Vt) {
  __shared__ __align__(16) bf16_t As[2][BM * BK];
  __shared__ __align__(16) bf16_t Bs[2][BN * BK];

  const int m0 = blockIdx.x * BM;
  const int n0 = blockIdx.y * BN;
  const int mode = blockIdx.z;
  const bf16_t* W = (mode == 0) ? Wq : (mode == 1) ? Wk : Wv;
  const float* bias = (mode == 0) ? bq : (mode == 1) ? bk : bv;

  GemmAcc acc = gemm_core(X, W, m0, n0, As, Bs);

  const int lane = threadIdx.x & 31;
  const int wave = threadIdx.x >> 5;
  const int mi = (wave & 3) * 32, ni = (wave >> 2) * 32;
  const int col = lane & 15, half = lane >> 4;

#pragma unroll
  for (int tm = 0; tm < 2; ++tm) {
#pragma unroll
    for (int tn = 0; tn < 2; ++tn) {
      const int o = n0 + ni + tn * 16 + col;
      const float bo = bias[o];
      const int head = o >> 6, d = o & (HD_ - 1);
#pragma unroll
      for (int r = 0; r < 8; ++r) {
        const int t  = m0 + mi + tm * 16 + r + 8 * half;
        const int bb = t >> 11;           // / S_
        const int s  = t & (S_ - 1);
        const float v = acc.a[tm][tn][r] + bo;
        if (mode == 2) {
          Vt[((size_t)(bb * NH_ + head) * HD_ + d) * S_ + s] = (bf16_t)v;
        } else {
          bf16_t* dst = (mode == 0) ? Q : K;
          dst[((size_t)(bb * NH_ + head) * S_ + s) * HD_ + d] = (bf16_t)v;
        }
      }
    }
  }
}

// ---------------------------------------------------------------------------
// Flash-style attention: 128-thread blocks (4 waves), each wave owns one
// 16-query tile; K/V chunks staged once per block into LDS (async, double-
// buffered) and shared by all 4 waves. grid: (S/64, NH, B).
// ---------------------------------------------------------------------------

// 4 async ops per thread per chunk: 2 for K (32x64), 2 for Vt (64x32)
__device__ __forceinline__ void attn_stage(const bf16_t* __restrict__ Kh,
                                           const bf16_t* __restrict__ Vth,
                                           int kc, bf16_t* Ks, bf16_t* Vs, int tid) {
#pragma unroll
  for (int i = 0; i < 2; ++i) {
    int f = (i * 128 + tid) * 8;            // 0..2047
    int row = f >> 6, col = f & 63;
    async_copy_16B(&Ks[f], Kh + (size_t)(kc + row) * HD_ + col);
  }
#pragma unroll
  for (int i = 0; i < 2; ++i) {
    int f = (i * 128 + tid) * 8;
    int row = f >> 5, col = f & 31;
    async_copy_16B(&Vs[f], Vth + (size_t)row * S_ + kc + col);
  }
}

__global__ void attn_kernel(const bf16_t* __restrict__ Q,
                            const bf16_t* __restrict__ K,
                            const bf16_t* __restrict__ Vt,
                            bf16_t* __restrict__ ctx) {
  const int tid  = threadIdx.x;
  const int lane = tid & 31;
  const int wave = tid >> 5;
  const int q0 = (blockIdx.x * 4 + wave) * 16;
  const int h  = blockIdx.y;
  const int b  = blockIdx.z;
  const size_t hb = (size_t)(b * NH_ + h);
  const bf16_t* Qh  = Q  + hb * S_ * HD_;
  const bf16_t* Kh  = K  + hb * S_ * HD_;
  const bf16_t* Vth = Vt + hb * HD_ * S_;

  __shared__ __align__(16) bf16_t Ks[2][32 * 64];   // K chunk (rows kc..kc+31)
  __shared__ __align__(16) bf16_t Vs[2][64 * 32];   // Vt chunk (64 d-rows x 32 keys)
  __shared__ __align__(16) bf16_t Pall[4][16 * 32]; // per-wave P tile
  bf16_t* P = Pall[wave];

  v16bf aq0 = load_A_frag(Qh, HD_, q0, 0, lane);
  v16bf aq1 = load_A_frag(Qh, HD_, q0, 32, lane);

  v8f acc0 = {}, acc1 = {}, acc2 = {}, acc3 = {};
  float mrun[8], lrun[8];
#pragma unroll
  for (int r = 0; r < 8; ++r) { mrun[r] = -3.0e38f; lrun[r] = 0.f; }
  const int col = lane & 15, half = lane >> 4;
  const float scale = 0.125f;   // 1/sqrt(64)

  attn_stage(Kh, Vth, 0, Ks[0], Vs[0], tid);

  for (int kc = 0; kc < S_; kc += 32) {
    const int cur = (kc >> 5) & 1;
    if (kc + 32 < S_) {
      attn_stage(Kh, Vth, kc + 32, Ks[cur ^ 1], Vs[cur ^ 1], tid);
      wait_async<4>();
    } else {
      wait_async<0>();
    }
    __syncthreads();

    const bf16_t* Kc = Ks[cur];
    const bf16_t* Vc = Vs[cur];

    // scores for keys [kc,kc+16) and [kc+16,kc+32) from LDS K tile
    v8f s0 = {}, s1 = {};
    {
      v16bf b0 = load_B_rows(Kc, HD_, 0, 0, lane);
      v16bf b1 = load_B_rows(Kc, HD_, 0, 32, lane);
      s0 = wmma_bf16(aq0, b0, s0);
      s0 = wmma_bf16(aq1, b1, s0);
    }
    {
      v16bf b0 = load_B_rows(Kc, HD_, 16, 0, lane);
      v16bf b1 = load_B_rows(Kc, HD_, 16, 32, lane);
      s1 = wmma_bf16(aq0, b0, s1);
      s1 = wmma_bf16(aq1, b1, s1);
    }

    // online softmax; C layout: lane = column, VGPR r = row r + 8*half
#pragma unroll
    for (int r = 0; r < 8; ++r) {
      float v0 = s0[r] * scale;
      float v1 = s1[r] * scale;
      float mx = fmaxf(v0, v1);
#pragma unroll
      for (int off = 1; off < 16; off <<= 1)
        mx = fmaxf(mx, __shfl_xor(mx, off, 32));
      float mnew = fmaxf(mrun[r], mx);
      float corr = __expf(mrun[r] - mnew);
      float p0 = __expf(v0 - mnew);
      float p1 = __expf(v1 - mnew);
      float rs = p0 + p1;
#pragma unroll
      for (int off = 1; off < 16; off <<= 1)
        rs += __shfl_xor(rs, off, 32);
      lrun[r] = lrun[r] * corr + rs;
      mrun[r] = mnew;
      acc0[r] *= corr; acc1[r] *= corr; acc2[r] *= corr; acc3[r] *= corr;
      P[(r + 8 * half) * 32 + col]      = (bf16_t)p0;
      P[(r + 8 * half) * 32 + 16 + col] = (bf16_t)p1;
    }
    // per-wave LDS RAW on P: DS ops are in-order within a wave
    v16bf ap  = load_A_frag(P, 32, 0, 0, lane);
    v16bf bv0 = load_B_rows(Vc, 32,  0, 0, lane);
    v16bf bv1 = load_B_rows(Vc, 32, 16, 0, lane);
    v16bf bv2 = load_B_rows(Vc, 32, 32, 0, lane);
    v16bf bv3 = load_B_rows(Vc, 32, 48, 0, lane);
    acc0 = wmma_bf16(ap, bv0, acc0);
    acc1 = wmma_bf16(ap, bv1, acc1);
    acc2 = wmma_bf16(ap, bv2, acc2);
    acc3 = wmma_bf16(ap, bv3, acc3);
    __syncthreads();            // protect K/V buffer reuse next iteration
  }

#pragma unroll
  for (int r = 0; r < 8; ++r) {
    const float inv = 1.0f / lrun[r];
    const int row = q0 + r + 8 * half;
    bf16_t* dst = ctx + ((size_t)(b * S_ + row)) * H_ + h * HD_;
    dst[ 0 + col] = (bf16_t)(acc0[r] * inv);
    dst[16 + col] = (bf16_t)(acc1[r] * inv);
    dst[32 + col] = (bf16_t)(acc2[r] * inv);
    dst[48 + col] = (bf16_t)(acc3[r] * inv);
  }
}

// ---------------------------------------------------------------------------
// Output dense + bias + residual -> f32. grid: (MTOK/BM, H/BN), block 256.
// ---------------------------------------------------------------------------
__global__ void out_gemm_kernel(const bf16_t* __restrict__ Ctx,
                                const bf16_t* __restrict__ Wd,
                                const float* __restrict__ bd,
                                const float* __restrict__ hs,
                                float* __restrict__ xbuf) {
  __shared__ __align__(16) bf16_t As[2][BM * BK];
  __shared__ __align__(16) bf16_t Bs[2][BN * BK];

  const int m0 = blockIdx.x * BM;
  const int n0 = blockIdx.y * BN;

  GemmAcc acc = gemm_core(Ctx, Wd, m0, n0, As, Bs);

  const int lane = threadIdx.x & 31;
  const int wave = threadIdx.x >> 5;
  const int mi = (wave & 3) * 32, ni = (wave >> 2) * 32;
  const int col = lane & 15, half = lane >> 4;

#pragma unroll
  for (int tm = 0; tm < 2; ++tm) {
#pragma unroll
    for (int tn = 0; tn < 2; ++tn) {
      const int o = n0 + ni + tn * 16 + col;
      const float bo = bd[o];
#pragma unroll
      for (int r = 0; r < 8; ++r) {
        const int t = m0 + mi + tm * 16 + r + 8 * half;
        xbuf[(size_t)t * H_ + o] = acc.a[tm][tn][r] + bo + hs[(size_t)t * H_ + o];
      }
    }
  }
}

// ---------------------------------------------------------------------------
// Row-wise LayerNorm: one 256-thread block per token row.
// ---------------------------------------------------------------------------
__global__ void ln_kernel(const float* __restrict__ x,
                          const float* __restrict__ gamma,
                          const float* __restrict__ beta,
                          float* __restrict__ out) {
  const int row = blockIdx.x;
  const int tid = threadIdx.x;
  const float* xr = x + (size_t)row * H_;

  float s = 0.f, ss = 0.f;
  for (int i = tid; i < H_; i += 256) {
    float v = xr[i];
    s += v; ss += v * v;
  }
  __shared__ float red0[256];
  __shared__ float red1[256];
  red0[tid] = s; red1[tid] = ss;
  __syncthreads();
  for (int off = 128; off > 0; off >>= 1) {
    if (tid < off) { red0[tid] += red0[tid + off]; red1[tid] += red1[tid + off]; }
    __syncthreads();
  }
  const float mean = red0[0] * (1.0f / H_);
  const float var  = red1[0] * (1.0f / H_) - mean * mean;
  const float inv  = rsqrtf(var + 1e-5f);
  for (int i = tid; i < H_; i += 256) {
    out[(size_t)row * H_ + i] = (xr[i] - mean) * inv * gamma[i] + beta[i];
  }
}

// ---------------------------------------------------------------------------
// Launch
// ---------------------------------------------------------------------------
extern "C" void kernel_launch(void* const* d_in, const int* in_sizes, int n_in,
                              void* d_out, int out_size, void* d_ws, size_t ws_size,
                              hipStream_t stream) {
  const float* hs = (const float*)d_in[0];
  const float* Wq = (const float*)d_in[1];
  const float* bq = (const float*)d_in[2];
  const float* Wk = (const float*)d_in[3];
  const float* bk = (const float*)d_in[4];
  const float* Wv = (const float*)d_in[5];
  const float* bv = (const float*)d_in[6];
  const float* Wd = (const float*)d_in[7];
  const float* bd = (const float*)d_in[8];
  const float* gamma = (const float*)d_in[9];
  const float* beta  = (const float*)d_in[10];
  float* out = (float*)d_out;

  // workspace carve-out (64 MB total)
  char* ws = (char*)d_ws;
  size_t off = 0;
  auto carve = [&](size_t bytes) { char* p = ws + off; off += bytes; return p; };
  bf16_t* Xbf  = (bf16_t*)carve((size_t)MTOK * H_ * sizeof(bf16_t));
  bf16_t* Wqb  = (bf16_t*)carve((size_t)H_ * H_ * sizeof(bf16_t));
  bf16_t* Wkb  = (bf16_t*)carve((size_t)H_ * H_ * sizeof(bf16_t));
  bf16_t* Wvb  = (bf16_t*)carve((size_t)H_ * H_ * sizeof(bf16_t));
  bf16_t* Wdb  = (bf16_t*)carve((size_t)H_ * H_ * sizeof(bf16_t));
  bf16_t* Qb   = (bf16_t*)carve((size_t)MTOK * H_ * sizeof(bf16_t));
  bf16_t* Kb   = (bf16_t*)carve((size_t)MTOK * H_ * sizeof(bf16_t));
  bf16_t* Vtb  = (bf16_t*)carve((size_t)MTOK * H_ * sizeof(bf16_t));
  bf16_t* Ctxb = (bf16_t*)carve((size_t)MTOK * H_ * sizeof(bf16_t));
  float*  xbuf = (float*)carve((size_t)MTOK * H_ * sizeof(float));
  (void)ws_size; (void)in_sizes; (void)n_in; (void)out_size;

  const int nX = MTOK * H_;
  const int nW = H_ * H_;
  cvt_kernel<<<(nX + 255) / 256, 256, 0, stream>>>(hs, Xbf, nX);
  cvt_kernel<<<(nW + 255) / 256, 256, 0, stream>>>(Wq, Wqb, nW);
  cvt_kernel<<<(nW + 255) / 256, 256, 0, stream>>>(Wk, Wkb, nW);
  cvt_kernel<<<(nW + 255) / 256, 256, 0, stream>>>(Wv, Wvb, nW);
  cvt_kernel<<<(nW + 255) / 256, 256, 0, stream>>>(Wd, Wdb, nW);

  qkv_gemm_kernel<<<dim3(MTOK / BM, H_ / BN, 3), 256, 0, stream>>>(
      Xbf, Wqb, Wkb, Wvb, bq, bk, bv, Qb, Kb, Vtb);

  attn_kernel<<<dim3(S_ / 64, NH_, B_), 128, 0, stream>>>(Qb, Kb, Vtb, Ctxb);

  out_gemm_kernel<<<dim3(MTOK / BM, H_ / BN), 256, 0, stream>>>(
      Ctxb, Wdb, bd, hs, xbuf);

  ln_kernel<<<MTOK, 256, 0, stream>>>(xbuf, gamma, beta, out);
}
